// GCLSTMCell_41979010351785
// MI455X (gfx1250) — matrix operand order repaired and checked by reading, not compiled, not run on hardware
//
#include <hip/hip_runtime.h>
#include <hip/hip_bf16.h>
#include <math.h>

#define HD 128          // hidden size
#define BN_EPS 1e-5f

typedef __attribute__((ext_vector_type(16))) __bf16 v16bf;
typedef __attribute__((ext_vector_type(16))) unsigned short v16us;
typedef __attribute__((ext_vector_type(8)))  float v8f;

union BF16x16 { v16us u; v16bf b; };

// float -> bf16 round-to-nearest-even (bit math; only used in the small
// one-shot packing kernels and the per-element A staging, never in GEMM loops)
__device__ __forceinline__ unsigned short f32_to_bf16_bits(float f) {
    unsigned int u = __float_as_uint(f);
    unsigned int r = u + 0x7fffu + ((u >> 16) & 1u);
    return (unsigned short)(r >> 16);
}

// ---------------------------------------------------------------------------
// Fragment layout maps (ISA 7.12.2, 16-bit operands, wave32)
//
// A 16x32 tile:  lane = hlf*16 + m  (m = row 0..15)
//   element j (0..15) holds K = base + hlf*8 + j', with j<8 -> base 0, j>=8 -> base 16
//   i.e. forward:  koff = (j<8 ? 0:16) + hlf*8 + (j&7)
//   inverse (q = K&31): hlf = (q>>3)&1 ; j = (q&7) + (q&16 ? 8 : 0)
//
// B 32x16 tile:  lane = kg*16 + n ; element j holds K = (j<8 ? 0:16) + kg*8 + (j&7)
// C/D 16x16:     lane = hlf*16 + n ; VGPR r holds row m = hlf*8 + r
// Packed storage: frag[(tileIdx*32 + lane)*16 + j] -> one contiguous 32B
// vector per lane per fragment.
// ---------------------------------------------------------------------------

// ---------------------------------------------------------------------------
// Kernel 0: zero the segment-sum accumulator
// ---------------------------------------------------------------------------
__global__ void zero_f32(float* __restrict__ p, size_t n) {
    size_t i = (size_t)blockIdx.x * blockDim.x + threadIdx.x;
    size_t stride = (size_t)gridDim.x * blockDim.x;
    for (; i < n; i += stride) p[i] = 0.0f;
}

// ---------------------------------------------------------------------------
// Pack x (N,128) fp32 -> bf16 A-fragment layout [tile][kk(4)][lane][j]
// ---------------------------------------------------------------------------
__global__ void pack_x_a(const float* __restrict__ x,
                         unsigned short* __restrict__ xp, size_t total) {
    size_t i = (size_t)blockIdx.x * blockDim.x + threadIdx.x;
    size_t stride = (size_t)gridDim.x * blockDim.x;
    for (; i < total; i += stride) {
        int row = (int)(i >> 7);
        int c   = (int)(i & 127);
        int tile = row >> 4, m = row & 15;
        int kk = c >> 5, q = c & 31;
        int hlf = (q >> 3) & 1;
        int j = (q & 7) + ((q & 16) ? 8 : 0);
        int lane = hlf * 16 + m;
        xp[(((size_t)tile * 4 + kk) * 32 + lane) * 16 + j] = f32_to_bf16_bits(x[i]);
    }
}

// ---------------------------------------------------------------------------
// Pack gcn_weight (K=128, N=128 row-major) -> B-fragment layout [kk(4)][nt(8)][lane][j]
// ---------------------------------------------------------------------------
__global__ void pack_gcn_b(const float* __restrict__ gw,
                           unsigned short* __restrict__ gwp) {
    int i = blockIdx.x * blockDim.x + threadIdx.x;   // 128*128 = 16384 threads
    if (i >= 128 * 128) return;
    int k = i >> 7, n = i & 127;
    int ktile = k >> 5, kg = (k >> 3) & 1, j = (k & 7) + ((k & 16) ? 8 : 0);
    int ntile = n >> 4, lane = kg * 16 + (n & 15);
    gwp[((ktile * 8 + ntile) * 32 + lane) * 16 + j] = f32_to_bf16_bits(gw[i]);
}

// ---------------------------------------------------------------------------
// Pack gate weights: B_cat[k][n], k=0..255 (x2h then h2h), n=0..511
// (x2h_w/h2h_w are (512,128) row-major, used transposed)
// -> B-fragment layout [kk(8)][ntile(32)][lane][j]
// ---------------------------------------------------------------------------
__global__ void pack_gates_b(const float* __restrict__ x2h_w,
                             const float* __restrict__ h2h_w,
                             unsigned short* __restrict__ wp) {
    int i = blockIdx.x * blockDim.x + threadIdx.x;   // 256*512 = 131072 threads
    if (i >= 256 * 512) return;
    int k = i >> 9, n = i & 511;
    float v = (k < 128) ? x2h_w[(size_t)n * HD + k]
                        : h2h_w[(size_t)n * HD + (k - 128)];
    int ktile = k >> 5, kg = (k >> 3) & 1, j = (k & 7) + ((k & 16) ? 8 : 0);
    int ntile = n >> 4, lane = kg * 16 + (n & 15);
    wp[(((size_t)ktile * 32 + ntile) * 32 + lane) * 16 + j] = f32_to_bf16_bits(v);
}

// ---------------------------------------------------------------------------
// Kernel 1: support = x @ gcn_weight  via bf16 WMMA on pre-packed fragments.
// One wave per 16-row strip; every fragment is one contiguous 32B lane load.
// ---------------------------------------------------------------------------
__global__ __launch_bounds__(128) void gcn_support_wmma(
    const unsigned short* __restrict__ xp,   // [tile][kk][lane][j]
    const unsigned short* __restrict__ gwp,  // [kk][nt][lane][j]
    float* __restrict__ support, int ntiles) {
    int wave = blockIdx.x * (blockDim.x >> 5) + (threadIdx.x >> 5);
    if (wave >= ntiles) return;          // wave-uniform guard: EXEC all-ones
    int lane = threadIdx.x & 31;
    int hlf  = lane >> 4;
    int n    = lane & 15;
    int row0 = wave * 16;

    BF16x16 afr[4];
#pragma unroll
    for (int kk = 0; kk < 4; ++kk)
        afr[kk].u = *(const v16us*)(xp + (((size_t)wave * 4 + kk) * 32 + lane) * 16);

#pragma unroll
    for (int nt = 0; nt < 8; ++nt) {
        v8f acc = {};
#pragma unroll
        for (int kk = 0; kk < 4; ++kk) {
            BF16x16 bfr;
            bfr.u = *(const v16us*)(gwp + ((size_t)(kk * 8 + nt) * 32 + lane) * 16);
            acc = __builtin_amdgcn_wmma_f32_16x16x32_bf16(
                false, afr[kk].b, false, bfr.b, (short)0, acc, false, false);
        }
#pragma unroll
        for (int r = 0; r < 8; ++r) {
            int mr = hlf * 8 + r;
            support[(size_t)(row0 + mr) * HD + nt * 16 + n] = acc[r];
        }
    }
}

// ---------------------------------------------------------------------------
// Kernel 2: agg[dst] += w * support[src]   (one wave per edge, float4 gather,
// global_atomic_add_f32 scatter; support/agg are L2-resident)
// ---------------------------------------------------------------------------
__global__ void spmm_scatter(const float* __restrict__ support,
                             const int* __restrict__ esrc,
                             const int* __restrict__ edst,
                             const float* __restrict__ ew,
                             float* __restrict__ agg, int E) {
    long long tid = (long long)blockIdx.x * blockDim.x + threadIdx.x;
    int e = (int)(tid >> 5);
    if (e >= E) return;
    int lane = (int)(tid & 31);
    int s = esrc[e], d = edst[e];
    float w = ew[e];
    const float4* sp = (const float4*)(support + (size_t)s * HD);
    float4 v = sp[lane];
    float* ap = agg + (size_t)d * HD + lane * 4;
    atomicAdd(ap + 0, w * v.x);
    atomicAdd(ap + 1, w * v.y);
    atomicAdd(ap + 2, w * v.z);
    atomicAdd(ap + 3, w * v.w);
}

// ---------------------------------------------------------------------------
// Kernel 3: fused  h = BN(relu(agg)+bias);  gates = [h|hx] @ B_cat + biases;
//           LSTM elementwise -> hy, cy.
// Block = 128 threads (4 waves).  A staged in LDS *in fragment order*:
// As[(kk*32+lane)*16 + j], 8KB.  Wave g computes gate g's 16x128 block with
// 8x8 v_wmma (K=256); gate blocks exchanged via LDS for the LSTM math.
// ---------------------------------------------------------------------------
__global__ __launch_bounds__(128) void gates_lstm_wmma(
    const float* __restrict__ agg, const float* __restrict__ hx,
    const float* __restrict__ cx,  const float* __restrict__ bias,
    const unsigned short* __restrict__ wp,   // packed B_cat [kk(8)][ntile(32)][lane][j]
    const float* __restrict__ x2h_b, const float* __restrict__ h2h_b,
    const float* __restrict__ gamma, const float* __restrict__ beta,
    const float* __restrict__ mean,  const float* __restrict__ var,
    float* __restrict__ hy, float* __restrict__ cy) {
    __shared__ float smem[4 * 16 * HD];            // 32 KB gate buffer
    unsigned short* As = (unsigned short*)smem;    // first 8 KB: packed A fragments

    int t = threadIdx.x;
    int row0 = blockIdx.x * 16;

    // Stage A = [ BN(relu(agg)+bias) | hx ] as bf16, directly in fragment order
    for (int idx = t; idx < 16 * 256; idx += 128) {
        int r = idx >> 8;          // row 0..15
        int c = idx & 255;         // K index 0..255
        float v;
        if (c < HD) {
            float a = agg[(size_t)(row0 + r) * HD + c];
            a = fmaxf(a, 0.0f) + bias[c];
            v = (a - mean[c]) * rsqrtf(var[c] + BN_EPS) * gamma[c] + beta[c];
        } else {
            v = hx[(size_t)(row0 + r) * HD + (c - HD)];
        }
        int kk = c >> 5, q = c & 31;
        int hlf = (q >> 3) & 1;
        int j = (q & 7) + ((q & 16) ? 8 : 0);
        int lane = hlf * 16 + r;
        As[((kk * 32 + lane) << 4) + j] = f32_to_bf16_bits(v);
    }
    __syncthreads();

    int lane = t & 31;
    int gate = t >> 5;          // 0:i 1:f 2:g 3:o
    int hlf  = lane >> 4;
    int n    = lane & 15;

    v8f acc[8];
#pragma unroll
    for (int nt = 0; nt < 8; ++nt) { v8f z = {}; acc[nt] = z; }

#pragma unroll
    for (int kk = 0; kk < 8; ++kk) {                         // K = 256
        BF16x16 afr;
        afr.u = *(const v16us*)&As[(kk * 32 + lane) << 4];
#pragma unroll
        for (int nt = 0; nt < 8; ++nt) {
            BF16x16 bfr;
            bfr.u = *(const v16us*)(wp + (((size_t)kk * 32 + gate * 8 + nt) * 32 + lane) * 16);
            acc[nt] = __builtin_amdgcn_wmma_f32_16x16x32_bf16(
                false, afr.b, false, bfr.b, (short)0, acc[nt], false, false);
        }
    }
    __syncthreads();   // all waves finished reading As; gate buffer may overwrite it

    // bias add + spill this wave's gate block to LDS (fp32)
#pragma unroll
    for (int nt = 0; nt < 8; ++nt) {
        int ng = gate * HD + nt * 16 + n;
        float gb = x2h_b[ng] + h2h_b[ng];
#pragma unroll
        for (int r = 0; r < 8; ++r) {
            int mr = hlf * 8 + r;
            smem[gate * (16 * HD) + mr * HD + nt * 16 + n] = acc[nt][r] + gb;
        }
    }
    __syncthreads();

    // LSTM elementwise: recombine i/f/g/o across waves via LDS
    for (int idx = t; idx < 16 * HD; idx += 128) {
        int r = idx >> 7;
        int c = idx & (HD - 1);
        float ig = smem[0 * 2048 + idx];
        float fg = smem[1 * 2048 + idx];
        float gg = smem[2 * 2048 + idx];
        float og = smem[3 * 2048 + idx];
        size_t off = (size_t)(row0 + r) * HD + c;
        float cprev = cx[off];
        float si = 1.0f / (1.0f + expf(-ig));
        float sf = 1.0f / (1.0f + expf(-fg));
        float so = 1.0f / (1.0f + expf(-og));
        float cnew = cprev * sf + si * tanhf(gg);
        cy[off] = cnew;
        hy[off] = so * tanhf(cnew);
    }
}

// ---------------------------------------------------------------------------
extern "C" void kernel_launch(void* const* d_in, const int* in_sizes, int n_in,
                              void* d_out, int out_size, void* d_ws, size_t ws_size,
                              hipStream_t stream) {
    const float* x      = (const float*)d_in[0];
    const float* hx     = (const float*)d_in[1];
    const float* cx     = (const float*)d_in[2];
    const int*   esrc   = (const int*)d_in[3];
    const int*   edst   = (const int*)d_in[4];
    const float* ew     = (const float*)d_in[5];
    const float* gw     = (const float*)d_in[6];
    const float* bias   = (const float*)d_in[7];
    const float* x2h_w  = (const float*)d_in[8];
    const float* x2h_b  = (const float*)d_in[9];
    const float* h2h_w  = (const float*)d_in[10];
    const float* h2h_b  = (const float*)d_in[11];
    const float* gamma  = (const float*)d_in[12];
    const float* beta   = (const float*)d_in[13];
    const float* mean   = (const float*)d_in[14];
    const float* var    = (const float*)d_in[15];

    const int N = in_sizes[0] / HD;      // 50000
    const int E = in_sizes[3];           // 800000
    const int ntiles = N / 16;           // 3125 (N is a multiple of 16)

    // Workspace layout (all offsets 32B aligned):
    //   support : N*H fp32                (25.6 MB)
    //   agg     : N*H fp32                (25.6 MB)
    //   xp      : N*H bf16 packed A       (12.8 MB)
    //   gwp     : 128*128 bf16 packed B   (32 KB)
    //   wp      : 256*512 bf16 packed B   (256 KB)
    float* support = (float*)d_ws;
    float* agg     = support + (size_t)N * HD;
    unsigned short* xp  = (unsigned short*)(agg + (size_t)N * HD);
    unsigned short* gwp = xp + (size_t)N * HD;
    unsigned short* wp  = gwp + 128 * 128;

    // 0) clear accumulator (ws is poisoned; atomics need zeros every call)
    zero_f32<<<2048, 256, 0, stream>>>(agg, (size_t)N * HD);

    // 0b) pack operands to bf16 fragment layouts (one-shot, amortized)
    {
        size_t total = (size_t)N * HD;
        pack_x_a<<<4096, 256, 0, stream>>>(x, xp, total);
        pack_gcn_b<<<(128 * 128 + 255) / 256, 256, 0, stream>>>(gw, gwp);
        pack_gates_b<<<(256 * 512 + 255) / 256, 256, 0, stream>>>(x2h_w, h2h_w, wp);
    }

    // 1) dense GCN GEMM via bf16 WMMA on packed fragments
    {
        int blocks = (ntiles + 3) / 4;   // 4 waves per 128-thread block
        gcn_support_wmma<<<blocks, 128, 0, stream>>>(xp, gwp, support, ntiles);
    }

    // 2) SpMM gather/scale/atomic-scatter (one wave per edge)
    {
        long long total = (long long)E * 32;
        int blocks = (int)((total + 255) / 256);
        spmm_scatter<<<blocks, 256, 0, stream>>>(support, esrc, edst, ew, agg, E);
    }

    // 3) fused BN + gates GEMM (WMMA) + LSTM elementwise
    gates_lstm_wmma<<<ntiles, 128, 0, stream>>>(agg, hx, cx, bias, wp,
                                                x2h_b, h2h_b,
                                                gamma, beta, mean, var,
                                                (float*)d_out,
                                                (float*)d_out + (size_t)N * HD);
}
